// GNN_46712064311359
// MI455X (gfx1250) — matrix operand order repaired and checked by reading, not compiled, not run on hardware
//
#include <hip/hip_runtime.h>
#include <hip/hip_bf16.h>

typedef __attribute__((ext_vector_type(16))) _Float16 v16h;
typedef __attribute__((ext_vector_type(8)))  float    v8f;

#define HP 128   // padded channel stride for h/e/tmp buffers (H=100 -> 128)

constexpr int AM_PLAIN  = 0;  // A[m][k]
constexpr int AM_SUM2   = 1;  // A0[m][k] + A1[m][k]           (z = h + aggr)
constexpr int AM_CONCAT = 2;  // [h[src[m]] | h[dst[m]] | e[m]] (K = 300, pad 320)
constexpr int EM_STORE   = 0; // out = acc + bias
constexpr int EM_RELU    = 1; // out = relu(acc + bias)
constexpr int EM_RESHALF = 2; // out = R[o] + 0.5*(acc + bias)

// Pack weight [Kact x Nact] (row-major, f32) into WMMA B-fragment order, f16:
//   idx = ((t*Kblocks + kblk)*32 + lane)*16 + j ; K = kblk*32 + (j&7) + 16*(j>>3) + 8*(lane>>4),
//   n = t*16 + (lane&15). Zero-padded -> GEMM B loads are contiguous & unguarded.
__global__ void pack_w_kernel(const float* __restrict__ W, _Float16* __restrict__ Wp,
                              int Kact, int Nact, int Kblocks, int Ntiles)
{
    long i = (long)blockIdx.x * blockDim.x + threadIdx.x;
    long total = (long)Ntiles * Kblocks * 512;
    if (i >= total) return;
    int j    = (int)(i & 15);
    int lane = (int)((i >> 4) & 31);
    long rest = i >> 9;
    int kblk = (int)(rest % Kblocks);
    int t    = (int)(rest / Kblocks);
    int kw = kblk * 32 + (j & 7) + ((j >> 3) << 4) + ((lane >> 4) << 3);
    int n  = t * 16 + (lane & 15);
    float v = (kw < Kact && n < Nact) ? W[(long)kw * Nact + n] : 0.0f;
    Wp[i] = (_Float16)v;
}

// edge_attr [E,16] -> zero-padded [E,32] so GEMM A loads need no K guard
__global__ void pad_eattr_kernel(const float* __restrict__ ea, float* __restrict__ out, int E)
{
    long t = (long)blockIdx.x * blockDim.x + threadIdx.x;
    if (t >= (long)E * 32) return;
    int m = (int)(t >> 5);
    int c = (int)(t & 31);
    out[t] = (c < 16) ? ea[(long)m * 16 + c] : 0.0f;
}

// A-fragment K runs: quads at base, base+4, base+16, base+20 (base = kblk*32 + 8*half).
// All runs are 4-aligned; concat segment boundaries (100/200/300) are multiples of 4,
// so a float4 run never straddles a segment -> fully branchless.
template<int AMODE>
__device__ __forceinline__ void load_a_quads(float4 (&q)[4], int kblk, int kofs,
                                             const float* rA, const float* rB,
                                             const float* hs, const float* hd,
                                             const float* em)
{
    const int base = kblk * 32 + kofs;
#pragma unroll
    for (int i = 0; i < 4; ++i) {
        const int k0 = base + ((i & 1) << 2) + ((i >> 1) << 4);  // +0,+4,+16,+20
        if (AMODE == AM_PLAIN) {
            q[i] = *(const float4*)(rA + k0);
        } else if (AMODE == AM_SUM2) {
            float4 u = *(const float4*)(rA + k0);
            float4 w = *(const float4*)(rB + k0);
            q[i] = make_float4(u.x + w.x, u.y + w.y, u.z + w.z, u.w + w.w);
        } else {
            const float* p3 = (k0 < 100) ? (hs + k0)
                             : (k0 < 200) ? (hd + (k0 - 100))
                                          : (em + (k0 - 200)); // K>=300 hits e's zero pad
            q[i] = *(const float4*)p3;
        }
    }
}

__device__ __forceinline__ v16h quads_to_f16(const float4 (&q)[4])
{
    v16h a;
#pragma unroll
    for (int i = 0; i < 4; ++i) {
        a[4 * i + 0] = (_Float16)q[i].x;
        a[4 * i + 1] = (_Float16)q[i].y;
        a[4 * i + 2] = (_Float16)q[i].z;
        a[4 * i + 3] = (_Float16)q[i].w;
    }
    return a;
}

// One wave computes a full 16 x (NT*16) output strip. Packed weights are staged
// in LDS once per workgroup; B fragments are double-buffered in registers
// (ds_load for tile t+1 in flight while wmma(t) executes), A quads are
// double-buffered so global latency overlaps the WMMA burst.
template<int AMODE, int EMODE, int NT>
__global__ __launch_bounds__(256)
void gemm_wmma_kernel(const float* __restrict__ A0,
                      const float* __restrict__ A1,
                      const int*   __restrict__ srcIdx,
                      const int*   __restrict__ dstIdx,
                      const _Float16* __restrict__ Wp,  // packed fragments
                      const float* __restrict__ bias_p, // [Nact]
                      const float* __restrict__ R,      // residual base (EM_RESHALF)
                      float*       __restrict__ Out,
                      int M, int Nact, int ldA, int ldOut, int Kblocks, int Mtiles)
{
    extern __shared__ char smem_raw[];
    _Float16* smem = (_Float16*)smem_raw;

    // cooperative weight staging (before any wave exits)
    {
        const int nvec = NT * Kblocks * 64;   // 16B chunks: 512 f16 = 64 chunks per (t,kblk)
        const float4* srcv = (const float4*)Wp;
        float4* dstv = (float4*)smem;
        for (int i = threadIdx.x; i < nvec; i += 256) dstv[i] = srcv[i];
    }
    __syncthreads();

    const int wave = threadIdx.x >> 5;
    const int lane = threadIdx.x & 31;
    const int mtile = blockIdx.x * 8 + wave;
    if (mtile >= Mtiles) return;               // wave-uniform exit

    const int half = lane >> 4;
    const int lrow = lane & 15;
    const int m_a  = mtile * 16 + lrow;
    const int m_c  = (m_a < M) ? m_a : (M - 1);   // clamp: garbage rows discarded at store

    const float* rA = nullptr; const float* rB = nullptr;
    const float* hs = nullptr; const float* hd = nullptr; const float* em = nullptr;
    if (AMODE == AM_CONCAT) {
        const int s = srcIdx[m_c];
        const int d = dstIdx[m_c];
        hs = A0 + (long)s   * HP;
        hd = A0 + (long)d   * HP;
        em = A1 + (long)m_c * HP;
    } else {
        rA = A0 + (long)m_c * ldA;
        if (AMODE == AM_SUM2) rB = A1 + (long)m_c * ldA;
    }

    v8f acc[NT];
#pragma unroll
    for (int t = 0; t < NT; ++t)
#pragma unroll
        for (int j = 0; j < 8; ++j) acc[t][j] = 0.0f;

    const int kofs = half << 3;
    float4 cur[4], nxt[4];
    load_a_quads<AMODE>(cur, 0, kofs, rA, rB, hs, hd, em);

    for (int kblk = 0; kblk < Kblocks; ++kblk) {
        const int kpre = (kblk + 1 < Kblocks) ? (kblk + 1) : kblk;  // clamped prefetch
        load_a_quads<AMODE>(nxt, kpre, kofs, rA, rB, hs, hd, em);
        const v16h a = quads_to_f16(cur);

        const _Float16* wrow = smem + ((long)kblk * 32 + lane) * 16;
        v16h bcur = *(const v16h*)(wrow);                 // tile 0
#pragma unroll
        for (int t = 0; t < NT; ++t) {
            v16h bnxt = bcur;
            if (t + 1 < NT)                               // prefetch tile t+1 before wmma(t)
                bnxt = *(const v16h*)(wrow + (long)(t + 1) * Kblocks * 512);
            acc[t] = __builtin_amdgcn_wmma_f32_16x16x32_f16(
                false, a, false, bcur, (short)0, acc[t], false, false);
            bcur = bnxt;
        }
#pragma unroll
        for (int i = 0; i < 4; ++i) cur[i] = nxt[i];
    }

#pragma unroll
    for (int t = 0; t < NT; ++t) {
        const int n = t * 16 + lrow;
        const float bias = (n < Nact) ? bias_p[n] : 0.0f;
#pragma unroll
        for (int j = 0; j < 8; ++j) {
            const int m = mtile * 16 + j + (half << 3);   // C/D row = j + 8*half
            if (m < M) {
                float v = acc[t][j] + bias;
                if (EMODE == EM_RELU) v = v > 0.0f ? v : 0.0f;
                const long o = (long)m * ldOut + n;
                if (EMODE == EM_RESHALF) v = R[o] + 0.5f * v;
                Out[o] = v;
            }
        }
    }
}

__global__ void zero_f32_kernel(float* __restrict__ p, long n)
{
    long i = (long)blockIdx.x * blockDim.x + threadIdx.x;
    if (i < n) p[i] = 0.0f;
}

// msg = relu(h[src] + e); aggr[dst] += msg  (4 channels per thread, skip zeros)
__global__ void scatter_add_msg_kernel(const float* __restrict__ h,
                                       const float* __restrict__ e,
                                       const int*   __restrict__ src,
                                       const int*   __restrict__ dst,
                                       float*       __restrict__ aggr,
                                       int E)
{
    long t = (long)blockIdx.x * blockDim.x + threadIdx.x;
    long nWork = (long)E * 25;               // 100 channels / 4
    if (t >= nWork) return;
    int eid = (int)(t / 25);
    int c   = (int)(t % 25) * 4;
    int s = src[eid], dd = dst[eid];
#pragma unroll
    for (int i = 0; i < 4; ++i) {
        float v = h[(long)s * HP + c + i] + e[(long)eid * HP + c + i];
        if (v > 0.0f) atomicAdd(&aggr[(long)dd * HP + c + i], v);
    }
}

// one block per channel: batch mean + biased var -> mu[c], rsqrt(var+eps)
__global__ __launch_bounds__(256)
void bn_stats_kernel(const float* __restrict__ x, int Nrows,
                     float* __restrict__ mu_out, float* __restrict__ rs_out)
{
    const int c = blockIdx.x;
    float s = 0.0f, s2 = 0.0f;
    for (int m = threadIdx.x; m < Nrows; m += 256) {
        float v = x[(long)m * HP + c];
        s += v; s2 += v * v;
    }
    __shared__ float sh[256], sh2[256];
    sh[threadIdx.x] = s; sh2[threadIdx.x] = s2;
    __syncthreads();
    for (int off = 128; off > 0; off >>= 1) {
        if (threadIdx.x < off) {
            sh[threadIdx.x]  += sh[threadIdx.x + off];
            sh2[threadIdx.x] += sh2[threadIdx.x + off];
        }
        __syncthreads();
    }
    if (threadIdx.x == 0) {
        float mu  = sh[0] / (float)Nrows;
        float var = sh2[0] / (float)Nrows - mu * mu;
        if (var < 0.0f) var = 0.0f;
        mu_out[c] = mu;
        rs_out[c] = rsqrtf(var + 1e-5f);
    }
}

// h = (h + relu(gamma*(xres-mu)*rs + beta)) * 0.5   over [N,100]
__global__ void bn_update_h_kernel(float* __restrict__ h,
                                   const float* __restrict__ xres,
                                   const float* __restrict__ mu,
                                   const float* __restrict__ rs,
                                   const float* __restrict__ g,
                                   const float* __restrict__ b,
                                   int Nrows)
{
    long t = (long)blockIdx.x * blockDim.x + threadIdx.x;
    if (t >= (long)Nrows * 100) return;
    int m = (int)(t / 100);
    int c = (int)(t % 100);
    float bn = g[c] * (xres[(long)m * HP + c] - mu[c]) * rs[c] + b[c];
    bn = bn > 0.0f ? bn : 0.0f;
    long o = (long)m * HP + c;
    h[o] = (h[o] + bn) * 0.5f;
}

// final 25 -> 1 projection (tiny; plain VALU)
__global__ void mlp3_kernel(const float* __restrict__ in,  // ld = 32
                            const float* __restrict__ w,   // [25]
                            const float* __restrict__ b,   // [1]
                            float* __restrict__ out, int E)
{
    int m = blockIdx.x * blockDim.x + threadIdx.x;
    if (m >= E) return;
    float acc = b[0];
#pragma unroll
    for (int k = 0; k < 25; ++k) acc += in[(long)m * 32 + k] * w[k];
    out[m] = acc;
}

static inline int cdiv(long a, long b) { return (int)((a + b - 1) / b); }

extern "C" void kernel_launch(void* const* d_in, const int* in_sizes, int n_in,
                              void* d_out, int out_size, void* d_ws, size_t ws_size,
                              hipStream_t stream)
{
    const int E = in_sizes[1] / 2;        // edge_index [2,E]
    const int N = in_sizes[0] / 64;       // x [N,64]

    const float* x_in  = (const float*)d_in[0];
    const int*   ei    = (const int*)d_in[1];
    const int*   src   = ei;
    const int*   dst   = ei + E;
    const float* eattr = (const float*)d_in[2];
    const float* w_m1 = (const float*)d_in[27];
    const float* b_m1 = (const float*)d_in[28];
    const float* w_m2 = (const float*)d_in[29];
    const float* b_m2 = (const float*)d_in[30];
    const float* w_m3 = (const float*)d_in[31];
    const float* b_m3 = (const float*)d_in[32];

    // workspace layout (f32)
    float* ws      = (float*)d_ws;
    float* h       = ws;                         // [N,128]
    float* aggr    = h    + (long)N * HP;        // [N,128]
    float* tmp_n   = aggr + (long)N * HP;        // [N,128]
    float* xres    = tmp_n + (long)N * HP;       // [N,128]
    float* mu      = xres + (long)N * HP;        // [128]
    float* rs      = mu + 128;                   // [128]
    float* ea_pad  = rs + 128;                   // [E,32]
    float* e       = ea_pad + (long)E * 32;      // [E,128]
    float* tmp_e   = e + (long)E * HP;           // [E,128]; later mlp1 out (ld 64)
    float* m2out   = tmp_e + (long)E * 64;       // mlp2 out (ld 32), inside tmp_e region
    _Float16* wpk  = (_Float16*)(tmp_e + (long)E * HP);  // packed weights (~430 KB)

    // ---- pack all weights into WMMA fragment order (f16, zero-padded) ----
    long wofs = 0;
    auto pack = [&](const float* W, int Kact, int Nact, int Kblocks, int Ntiles) -> _Float16* {
        _Float16* dstp = wpk + wofs;
        long total = (long)Ntiles * Kblocks * 512;
        pack_w_kernel<<<cdiv(total, 256), 256, 0, stream>>>(W, dstp, Kact, Nact, Kblocks, Ntiles);
        wofs += total;
        return dstp;
    };
    _Float16* P_ne = pack((const float*)d_in[3], 64, 100, 2, 8);
    _Float16* P_ee = pack((const float*)d_in[5], 16, 100, 1, 8);
    _Float16* P_c1[2], *P_c2[2], *P_e1[2], *P_e2[2];
    for (int l = 0; l < 2; ++l) {
        P_c1[l] = pack((const float*)d_in[7 + 10 * l + 0], 100, 100, 4, 8);
        P_c2[l] = pack((const float*)d_in[7 + 10 * l + 2], 100, 100, 4, 8);
        P_e1[l] = pack((const float*)d_in[7 + 10 * l + 6], 300, 100, 10, 8);
        P_e2[l] = pack((const float*)d_in[7 + 10 * l + 8], 100, 100, 4, 8);
    }
    _Float16* P_m1 = pack(w_m1, 300, 50, 10, 4);
    _Float16* P_m2 = pack(w_m2, 50, 25, 2, 2);

    pad_eattr_kernel<<<cdiv((long)E * 32, 256), 256, 0, stream>>>(eattr, ea_pad, E);

#define LAUNCH_GEMM(AM, EM, NT, A0, A1, SRC, DST, WP, BIAS, RES, OUT, M_, NA_, LDA_, LDOUT_, KB_)   \
    do {                                                                                            \
        int mtiles_ = ((M_) + 15) / 16;                                                             \
        int blocks_ = (mtiles_ + 7) / 8;                                                            \
        size_t lds_ = (size_t)(NT) * (KB_) * 1024;                                                  \
        gemm_wmma_kernel<AM, EM, NT><<<blocks_, 256, lds_, stream>>>(                               \
            A0, A1, SRC, DST, WP, BIAS, RES, OUT, M_, NA_, LDA_, LDOUT_, KB_, mtiles_);             \
    } while (0)

    // h = x @ W_ne + b_ne  (padded cols written as exact zeros)
    LAUNCH_GEMM(AM_PLAIN, EM_STORE, 8, x_in, nullptr, nullptr, nullptr,
                P_ne, (const float*)d_in[4], nullptr, h, N, 100, 64, HP, 2);
    // e = edge_attr @ W_ee + b_ee
    LAUNCH_GEMM(AM_PLAIN, EM_STORE, 8, ea_pad, nullptr, nullptr, nullptr,
                P_ee, (const float*)d_in[6], nullptr, e, E, 100, 32, HP, 1);

    for (int l = 0; l < 2; ++l) {
        const float* cb1 = (const float*)d_in[7 + 10 * l + 1];
        const float* cb2 = (const float*)d_in[7 + 10 * l + 3];
        const float* bng = (const float*)d_in[7 + 10 * l + 4];
        const float* bnb = (const float*)d_in[7 + 10 * l + 5];
        const float* eb1 = (const float*)d_in[7 + 10 * l + 7];
        const float* eb2 = (const float*)d_in[7 + 10 * l + 9];

        // aggr = 0 ; aggr[dst] += relu(h[src] + e)
        zero_f32_kernel<<<cdiv((long)N * HP, 256), 256, 0, stream>>>(aggr, (long)N * HP);
        scatter_add_msg_kernel<<<cdiv((long)E * 25, 256), 256, 0, stream>>>(h, e, src, dst, aggr, E);

        // tmp_n = relu((h + aggr) @ W1 + b1) ; xres = tmp_n @ W2 + b2
        LAUNCH_GEMM(AM_SUM2, EM_RELU, 8, h, aggr, nullptr, nullptr,
                    P_c1[l], cb1, nullptr, tmp_n, N, 100, HP, HP, 4);
        LAUNCH_GEMM(AM_PLAIN, EM_STORE, 8, tmp_n, nullptr, nullptr, nullptr,
                    P_c2[l], cb2, nullptr, xres, N, 100, HP, HP, 4);

        // batch-norm stats + fused h update: h = (h + relu(bn(xres))) * 0.5
        bn_stats_kernel<<<100, 256, 0, stream>>>(xres, N, mu, rs);
        bn_update_h_kernel<<<cdiv((long)N * 100, 256), 256, 0, stream>>>(h, xres, mu, rs, bng, bnb, N);

        // tmp_e = relu([h_src|h_dst|e] @ Wem1 + bem1)   (gather fused into A-fragments)
        LAUNCH_GEMM(AM_CONCAT, EM_RELU, 8, h, e, src, dst,
                    P_e1[l], eb1, nullptr, tmp_e, E, 100, HP, HP, 10);
        // e = e + 0.5 * (tmp_e @ Wem2 + bem2)           (residual fused, in place)
        LAUNCH_GEMM(AM_PLAIN, EM_RESHALF, 8, tmp_e, nullptr, nullptr, nullptr,
                    P_e2[l], eb2, e, e, E, 100, HP, HP, 4);
    }

    // final edge MLP: relu(concat @ W1 + b1) -> relu(@ W2 + b2) -> @ W3 + b3
    LAUNCH_GEMM(AM_CONCAT, EM_RELU, 4, h, e, src, dst,
                P_m1, b_m1, nullptr, tmp_e, E, 50, HP, 64, 10);
    LAUNCH_GEMM(AM_PLAIN, EM_RELU, 2, tmp_e, nullptr, nullptr, nullptr,
                P_m2, b_m2, nullptr, m2out, E, 25, 64, 32, 2);
    mlp3_kernel<<<cdiv(E, 256), 256, 0, stream>>>(m2out, w_m3, b_m3, (float*)d_out, E);

#undef LAUNCH_GEMM
}
